// DebugBertLayer_87093346828840
// MI455X (gfx1250) — matrix operand order repaired and checked by reading, not compiled, not run on hardware
//
#include <hip/hip_runtime.h>
#include <cstdint>

typedef __attribute__((ext_vector_type(16))) _Float16 v16h;
typedef __attribute__((ext_vector_type(8)))  float    v8f;

#define D_MODEL 768
#define D_FF    3072
#define NROWS   8192      // B*S = 4*2048
#define LN_EPS  1e-12f

#define BM   128
#define BN   128
#define BK   32
#define LDSK 40           // padded LDS row stride in halves (80B, 16B-aligned)

// ---------------------------------------------------------------------------
// cvec[j] = 0.01 * rowsum(Wo)[j] + bo[j]   (attention collapses to a constant)
// ---------------------------------------------------------------------------
__global__ void cvec_kernel(const float* __restrict__ Wo, const float* __restrict__ bo,
                            float* __restrict__ cvec) {
    int j = blockIdx.x * blockDim.x + threadIdx.x;
    if (j >= D_MODEL) return;
    const float* row = Wo + (size_t)j * D_MODEL;
    float s = 0.f;
    for (int k = 0; k < D_MODEL; ++k) s += row[k];
    cvec[j] = 0.01f * s + bo[j];
}

__global__ void f32_to_f16_kernel(const float* __restrict__ in, _Float16* __restrict__ out, int n) {
    int i = blockIdx.x * blockDim.x + threadIdx.x;
    if (i < n) out[i] = (_Float16)in[i];
}

// ---------------------------------------------------------------------------
// Row LayerNorm over 768 columns, one wave32 per row (8 rows / 256-thread block)
// ADD_CVEC: add per-column constant before LN.  OUT_HALF: also emit f16 copy.
// ---------------------------------------------------------------------------
template<int ADD_CVEC, int OUT_HALF>
__global__ __launch_bounds__(256)
void ln_kernel(const float* __restrict__ x, const float* __restrict__ cvec,
               const float* __restrict__ gamma, const float* __restrict__ beta,
               float* __restrict__ outF, _Float16* __restrict__ outH) {
    const int row  = blockIdx.x * 8 + (threadIdx.x >> 5);
    const int lane = threadIdx.x & 31;
    const float* xr = x + (size_t)row * D_MODEL;

    float vals[24];
    float s = 0.f, ss = 0.f;
#pragma unroll
    for (int i = 0; i < 24; ++i) {
        const int c = lane + i * 32;
        float v = xr[c];
        if (ADD_CVEC) v += cvec[c];
        vals[i] = v;
        s += v; ss += v * v;
    }
#pragma unroll
    for (int off = 16; off > 0; off >>= 1) {
        s  += __shfl_xor(s,  off, 32);
        ss += __shfl_xor(ss, off, 32);
    }
    const float mu   = s * (1.0f / D_MODEL);
    const float var  = ss * (1.0f / D_MODEL) - mu * mu;
    const float rstd = rsqrtf(var + LN_EPS);
#pragma unroll
    for (int i = 0; i < 24; ++i) {
        const int c = lane + i * 32;
        const float y = (vals[i] - mu) * rstd * gamma[c] + beta[c];
        outF[(size_t)row * D_MODEL + c] = y;
        if (OUT_HALF) outH[(size_t)row * D_MODEL + c] = (_Float16)y;
    }
}

// ---------------------------------------------------------------------------
// WMMA GEMM:  C[M,N] = A[M,K] * W[N,K]^T  (both K-contiguous, f16, fp32 accum)
// Block tile 128x128, BK=32, 8 waves as 2(M)x4(N), wave tile 64x32.
// Double-buffered LDS staging; v_wmma_f32_16x16x32_f16 inner product.
// EPI==0: y = gelu_exact(acc + bias[n])            -> f16 (feeds next GEMM)
// EPI==1: y = acc + bias[n] + resid[m,n]           -> f32 (feeds LayerNorm)
// ---------------------------------------------------------------------------
template<int EPI>
__global__ __launch_bounds__(256)
void gemm_wmma_kernel(const _Float16* __restrict__ A,
                      const _Float16* __restrict__ W,
                      const float*    __restrict__ bias,
                      const float*    __restrict__ resid,
                      _Float16*       __restrict__ outH,
                      float*          __restrict__ outF,
                      int M, int N, int K) {
    __shared__ __align__(16) _Float16 As[2][BM * LDSK];
    __shared__ __align__(16) _Float16 Bs[2][BN * LDSK];

    const int tid  = threadIdx.x;
    const int lane = tid & 31;
    const int wave = tid >> 5;
    const int wm   = wave >> 2;     // 0..1
    const int wn   = wave & 3;      // 0..3
    const int l16  = lane & 15;
    const bool hi  = lane >= 16;

    const int blockM = blockIdx.y * BM;
    const int blockN = blockIdx.x * BN;
    const int nkt    = K / BK;

    auto load_tile = [&](int buf, int kt) {
        const int k0 = kt * BK;
#pragma unroll
        for (int i = 0; i < 2; ++i) {
            const int c   = tid + i * 256;     // 0..511 chunk id
            const int row = c >> 2;            // 0..127
            const int col = (c & 3) * 8;       // 0,8,16,24 halves
            const uint4 av = *(const uint4*)(A + (size_t)(blockM + row) * K + k0 + col);
            *(uint4*)(&As[buf][row * LDSK + col]) = av;
            const uint4 bv = *(const uint4*)(W + (size_t)(blockN + row) * K + k0 + col);
            *(uint4*)(&Bs[buf][row * LDSK + col]) = bv;
        }
        // L2 prefetch two k-tiles ahead (global_prefetch_b8)
        if (k0 + 2 * BK < K) {
            const int row = tid >> 2;
            const int col = (tid & 3) * 8;
            __builtin_prefetch(A + (size_t)(blockM + row) * K + k0 + 2 * BK + col, 0, 1);
            __builtin_prefetch(W + (size_t)(blockN + row) * K + k0 + 2 * BK + col, 0, 1);
        }
    };

    v8f acc[4][2] = {};

    load_tile(0, 0);
    __syncthreads();

    for (int kt = 0; kt < nkt; ++kt) {
        const int buf = kt & 1;
        if (kt + 1 < nkt) load_tile(buf ^ 1, kt + 1);

        const _Float16* as = As[buf];
        const _Float16* bs = Bs[buf];

        // A fragment (16x32, ISA 7.12.2): lane<16 -> K{0..7,16..23}, lane>=16 -> K{8..15,24..31}
        const int akb = hi ? 8 : 0;
        // B fragment (32x16): lane<16 -> K{0..7,16..23}, lane>=16 -> K{8..15,24..31}
        const int bkb = hi ? 8 : 0;

        v16h afrag[4], bfrag[2];
#pragma unroll
        for (int mi = 0; mi < 4; ++mi) {
            const _Float16* p = as + (wm * 64 + mi * 16 + l16) * LDSK + akb;
            ((uint4*)&afrag[mi])[0] = *(const uint4*)(p);
            ((uint4*)&afrag[mi])[1] = *(const uint4*)(p + 16);
        }
#pragma unroll
        for (int ni = 0; ni < 2; ++ni) {
            const _Float16* p = bs + (wn * 32 + ni * 16 + l16) * LDSK + bkb;
            ((uint4*)&bfrag[ni])[0] = *(const uint4*)(p);
            ((uint4*)&bfrag[ni])[1] = *(const uint4*)(p + 16);
        }
#pragma unroll
        for (int mi = 0; mi < 4; ++mi)
#pragma unroll
            for (int ni = 0; ni < 2; ++ni)
                acc[mi][ni] = __builtin_amdgcn_wmma_f32_16x16x32_f16(
                    false, afrag[mi], false, bfrag[ni], (short)0, acc[mi][ni], false, false);

        __syncthreads();
    }

    // Epilogue. C/D layout: lane<16: VGPR r -> (M=r, N=lane); lane>=16: (M=8+r, N=lane-16)
    const int rB = hi ? 8 : 0;
#pragma unroll
    for (int mi = 0; mi < 4; ++mi) {
#pragma unroll
        for (int ni = 0; ni < 2; ++ni) {
            const int col  = blockN + wn * 32 + ni * 16 + l16;
            const int row0 = blockM + wm * 64 + mi * 16 + rB;
            const float b  = bias[col];
#pragma unroll
            for (int r = 0; r < 8; ++r) {
                const int row = row0 + r;
                float x = acc[mi][ni][r] + b;
                if (EPI == 0) {
                    const float g = 0.5f * x * (1.0f + erff(x * 0.70710678118654752f));
                    outH[(size_t)row * N + col] = (_Float16)g;
                } else {
                    x += resid[(size_t)row * N + col];
                    outF[(size_t)row * N + col] = x;
                }
            }
        }
    }
}

// ---------------------------------------------------------------------------
extern "C" void kernel_launch(void* const* d_in, const int* in_sizes, int n_in,
                              void* d_out, int out_size, void* d_ws, size_t ws_size,
                              hipStream_t stream) {
    const float* hidden = (const float*)d_in[0];
    // d_in[1..6] (Wq,bq,Wk,bk,Wv,bv) are dead: q/k/v are overwritten by 0.01
    const float* Wo    = (const float*)d_in[7];
    const float* bo    = (const float*)d_in[8];
    const float* ln1_g = (const float*)d_in[9];
    const float* ln1_b = (const float*)d_in[10];
    const float* Wi    = (const float*)d_in[11];
    const float* bi    = (const float*)d_in[12];
    const float* Wf    = (const float*)d_in[13];
    const float* bf    = (const float*)d_in[14];
    const float* ln2_g = (const float*)d_in[15];
    const float* ln2_b = (const float*)d_in[16];

    char* ws = (char*)d_ws;
    size_t off = 0;
    auto alloc = [&](size_t bytes) -> void* {
        void* p = ws + off;
        off += (bytes + 255) & ~(size_t)255;
        return p;
    };
    float*    cvec     = (float*)   alloc((size_t)D_MODEL * 4);
    float*    attn_f32 = (float*)   alloc((size_t)NROWS * D_MODEL * 4);
    _Float16* attn_f16 = (_Float16*)alloc((size_t)NROWS * D_MODEL * 2);
    _Float16* Wi_h     = (_Float16*)alloc((size_t)D_FF * D_MODEL * 2);
    _Float16* Wf_h     = (_Float16*)alloc((size_t)D_MODEL * D_FF * 2);
    _Float16* h_f16    = (_Float16*)alloc((size_t)NROWS * D_FF * 2);
    float*    t_f32    = (float*)   alloc((size_t)NROWS * D_MODEL * 4);

    // 1) attention collapsed to per-column constant
    cvec_kernel<<<(D_MODEL + 255) / 256, 256, 0, stream>>>(Wo, bo, cvec);

    // 2) f16 weight copies for WMMA
    {
        const int n = D_FF * D_MODEL;
        f32_to_f16_kernel<<<(n + 255) / 256, 256, 0, stream>>>(Wi, Wi_h, n);
        f32_to_f16_kernel<<<(n + 255) / 256, 256, 0, stream>>>(Wf, Wf_h, n);
    }

    // 3) attn_out = LN(hidden + cvec)   (f32 for residual, f16 for GEMM A)
    ln_kernel<1, 1><<<NROWS / 8, 256, 0, stream>>>(hidden, cvec, ln1_g, ln1_b,
                                                   attn_f32, attn_f16);

    // 4) h = gelu(attn_out @ Wi^T + bi)   [8192 x 3072]
    gemm_wmma_kernel<0><<<dim3(D_FF / BN, NROWS / BM), 256, 0, stream>>>(
        attn_f16, Wi_h, bi, nullptr, h_f16, nullptr, NROWS, D_FF, D_MODEL);

    // 5) t = h @ Wf^T + bf + attn_out     [8192 x 768]
    gemm_wmma_kernel<1><<<dim3(D_MODEL / BN, NROWS / BM), 256, 0, stream>>>(
        h_f16, Wf_h, bf, attn_f32, nullptr, t_f32, NROWS, D_MODEL, D_FF);

    // 6) out = LN(t)
    ln_kernel<0, 0><<<NROWS / 8, 256, 0, stream>>>(t_f32, nullptr, ln2_g, ln2_b,
                                                   (float*)d_out, nullptr);
}
